// encoderLSTM_43834436223393
// MI455X (gfx1250) — matrix operand order repaired and checked by reading, not compiled, not run on hardware
//
#include <hip/hip_runtime.h>

// ---------------------------------------------------------------------------
// LSTM encoder for MI455X (gfx1250), bf16 WMMA (v_wmma_f32_16x16x32_bf16)
//   x:[128,512,512] f32, W_ih:[4096,512], W_hh:[4096,1024], b_ih/b_hh:[4096],
//   fc_w:[1024,1024], fc_b:[1024]  ->  out:[128,1024] f32
// ---------------------------------------------------------------------------

typedef __attribute__((ext_vector_type(16))) __bf16 v16bf;
typedef __attribute__((ext_vector_type(8)))  float  v8f;

#define D_IN   512
#define H_DIM  1024
#define G4     4096
#define BATCH  128
#define SEQ    512

__device__ __forceinline__ unsigned short f2bf(float f) {
    unsigned int u = __float_as_uint(f);
    u += 0x7FFFu + ((u >> 16) & 1u);        // round-to-nearest-even
    return (unsigned short)(u >> 16);
}

union Frag32 {                 // 32-byte WMMA operand (16 bf16 / 8 VGPRs)
    v16bf          bf;
    uint4          u[2];
    unsigned short s[16];
};

// ------------------------------- init kernels ------------------------------

// vectorized f32 -> bf16 (n must be a multiple of 8; all our sizes are)
__global__ void k_f32_to_bf16_v8(const float* __restrict__ src,
                                 unsigned int* __restrict__ dst, int n8) {
    int i = blockIdx.x * blockDim.x + threadIdx.x;
    int stride = gridDim.x * blockDim.x;
    for (; i < n8; i += stride) {
        float4 f0 = *(const float4*)(src + (size_t)i * 8);
        float4 f1 = *(const float4*)(src + (size_t)i * 8 + 4);
        uint4 o;
        o.x = (unsigned)f2bf(f0.x) | ((unsigned)f2bf(f0.y) << 16);
        o.y = (unsigned)f2bf(f0.z) | ((unsigned)f2bf(f0.w) << 16);
        o.z = (unsigned)f2bf(f1.x) | ((unsigned)f2bf(f1.y) << 16);
        o.w = (unsigned)f2bf(f1.z) | ((unsigned)f2bf(f1.w) << 16);
        *(uint4*)(dst + (size_t)i * 4) = o;
    }
}

__global__ void k_bias_zero(const float* __restrict__ b_ih,
                            const float* __restrict__ b_hh,
                            float* __restrict__ bsum,
                            float* __restrict__ c,
                            unsigned int* __restrict__ h0,
                            unsigned int* __restrict__ h1) {
    int i = blockIdx.x * blockDim.x + threadIdx.x;
    int stride = gridDim.x * blockDim.x;
    for (int j = i; j < G4; j += stride) bsum[j] = b_ih[j] + b_hh[j];
    for (int j = i; j < BATCH * H_DIM; j += stride) c[j] = 0.0f;
    for (int j = i; j < BATCH * H_DIM / 2; j += stride) { h0[j] = 0u; h1[j] = 0u; }
}

// ------------------------------- LSTM step ---------------------------------
// One launch per timestep. 64 blocks x 256 threads = 512 waves.
// wave -> (m_tile 0..7) x (gate-column strip c0 = 16*(wid>>3), 0..1008).
// Each wave accumulates the i,f,g,o 16x16 tiles for its (m_tile,c0) cell tile
// and applies the LSTM cell in registers.
//
// XBF16 = true : x was pre-converted to bf16 (pure b128-load + WMMA loops)
// XBF16 = false: convert x f32->bf16 on load (workspace-constrained fallback)

template <bool XBF16>
__global__ __launch_bounds__(256) void k_lstm_step(
    const float* __restrict__ x,              // [128,512,512] f32 (fallback)
    const unsigned short* __restrict__ x_bf,  // [128,512,512] bf16 (fast path)
    const unsigned short* __restrict__ wih,   // bf16 [4096,512] row-major
    const unsigned short* __restrict__ whh,   // bf16 [4096,1024] row-major
    const float* __restrict__ bsum,           // [4096]
    const unsigned short* __restrict__ h_in,  // bf16 [128,1024]
    unsigned short* __restrict__ h_out,       // bf16 [128,1024]
    float* __restrict__ c,                    // [128,1024]
    int t)
{
    const int lane   = threadIdx.x & 31;
    const int wave   = threadIdx.x >> 5;
    const int wid    = blockIdx.x * 8 + wave;     // 0..511
    const int m_tile = wid & 7;                   // 0..7  (rows 16*m_tile..+15)
    const int c0     = (wid >> 3) << 4;           // 0..1008 (cell column base)
    const int n      = lane & 15;                 // A row within tile / B,C col
    const int half   = lane >> 4;

    v8f acc[4];
    #pragma unroll
    for (int g = 0; g < 4; ++g)
        #pragma unroll
        for (int r = 0; r < 8; ++r) acc[g][r] = 0.0f;

    const int m_row = m_tile * 16 + n;            // 0..127

    // ---- K part 1: x[., t, .] ----
    if (XBF16) {
        const unsigned short* xrow = x_bf + ((size_t)m_row * SEQ + t) * D_IN;
        if (t + 1 < SEQ) __builtin_prefetch(xrow + D_IN, 0, 1);  // warm L2
        for (int k0 = 0; k0 < D_IN; k0 += 32) {
            Frag32 a;
            const unsigned short* p1 = xrow + k0 + 8 * half;
            a.u[0] = *(const uint4*)(p1);
            a.u[1] = *(const uint4*)(p1 + 16);
            #pragma unroll
            for (int g = 0; g < 4; ++g) {
                Frag32 b;   // B[k][col] = W[g*H + c0 + n][k] : contiguous in k
                const unsigned short* wp =
                    wih + (size_t)(g * H_DIM + c0 + n) * D_IN + k0 + 8 * half;
                b.u[0] = *(const uint4*)(wp);
                b.u[1] = *(const uint4*)(wp + 16);
                acc[g] = __builtin_amdgcn_wmma_f32_16x16x32_bf16(
                    false, a.bf, false, b.bf, (short)0, acc[g], false, false);
            }
        }
    } else {
        const float* xrow = x + ((size_t)m_row * SEQ + t) * D_IN;
        if (t + 1 < SEQ) __builtin_prefetch(xrow + D_IN, 0, 1);
        for (int k0 = 0; k0 < D_IN; k0 += 32) {
            Frag32 a;
            const float* p1 = xrow + k0 + 8 * half;   // K = k0+8h .. +7
            const float* p2 = p1 + 16;                // K = k0+16+8h .. +7
            float4 f0 = *(const float4*)(p1);
            float4 f1 = *(const float4*)(p1 + 4);
            float4 f2 = *(const float4*)(p2);
            float4 f3 = *(const float4*)(p2 + 4);
            a.s[0]=f2bf(f0.x); a.s[1]=f2bf(f0.y); a.s[2]=f2bf(f0.z); a.s[3]=f2bf(f0.w);
            a.s[4]=f2bf(f1.x); a.s[5]=f2bf(f1.y); a.s[6]=f2bf(f1.z); a.s[7]=f2bf(f1.w);
            a.s[8]=f2bf(f2.x); a.s[9]=f2bf(f2.y); a.s[10]=f2bf(f2.z); a.s[11]=f2bf(f2.w);
            a.s[12]=f2bf(f3.x); a.s[13]=f2bf(f3.y); a.s[14]=f2bf(f3.z); a.s[15]=f2bf(f3.w);
            #pragma unroll
            for (int g = 0; g < 4; ++g) {
                Frag32 b;
                const unsigned short* wp =
                    wih + (size_t)(g * H_DIM + c0 + n) * D_IN + k0 + 8 * half;
                b.u[0] = *(const uint4*)(wp);
                b.u[1] = *(const uint4*)(wp + 16);
                acc[g] = __builtin_amdgcn_wmma_f32_16x16x32_bf16(
                    false, a.bf, false, b.bf, (short)0, acc[g], false, false);
            }
        }
    }

    // ---- K part 2: h_in (bf16) ----
    const unsigned short* hrow = h_in + (size_t)m_row * H_DIM;
    for (int k0 = 0; k0 < H_DIM; k0 += 32) {
        Frag32 a;
        const unsigned short* p1 = hrow + k0 + 8 * half;
        a.u[0] = *(const uint4*)(p1);
        a.u[1] = *(const uint4*)(p1 + 16);

        #pragma unroll
        for (int g = 0; g < 4; ++g) {
            Frag32 b;
            const unsigned short* wp =
                whh + (size_t)(g * H_DIM + c0 + n) * H_DIM + k0 + 8 * half;
            b.u[0] = *(const uint4*)(wp);
            b.u[1] = *(const uint4*)(wp + 16);
            acc[g] = __builtin_amdgcn_wmma_f32_16x16x32_bf16(
                false, a.bf, false, b.bf, (short)0, acc[g], false, false);
        }
    }

    // ---- LSTM cell, fully in registers for this wave's tile ----
    const int col = c0 + n;                       // cell column 0..1023
    const float bi = bsum[col];
    const float bf = bsum[H_DIM + col];
    const float bg = bsum[2 * H_DIM + col];
    const float bo = bsum[3 * H_DIM + col];
    #pragma unroll
    for (int r = 0; r < 8; ++r) {
        const int m = m_tile * 16 + r + 8 * half; // C/D row mapping
        float iv = 1.0f / (1.0f + __expf(-(acc[0][r] + bi)));
        float fv = 1.0f / (1.0f + __expf(-(acc[1][r] + bf)));
        float gv = tanhf(acc[2][r] + bg);
        float ov = 1.0f / (1.0f + __expf(-(acc[3][r] + bo)));
        const size_t idx = (size_t)m * H_DIM + col;
        float cnew = fv * c[idx] + iv * gv;
        c[idx] = cnew;
        h_out[idx] = f2bf(ov * tanhf(cnew));
    }
}

// ------------------------------- final FC ----------------------------------
// out[128,1024] = h @ fc_w^T + fc_b, same WMMA scheme (512 waves, K=1024).

__global__ __launch_bounds__(256) void k_fc(
    const unsigned short* __restrict__ h_bf,   // bf16 [128,1024]
    const unsigned short* __restrict__ fcw,    // bf16 [1024,1024] row-major
    const float* __restrict__ fcb,             // [1024]
    float* __restrict__ out)                   // [128,1024]
{
    const int lane   = threadIdx.x & 31;
    const int wave   = threadIdx.x >> 5;
    const int wid    = blockIdx.x * 8 + wave;     // 0..511
    const int m_tile = wid & 7;
    const int n0     = (wid >> 3) << 4;           // 0..1008
    const int n      = lane & 15;
    const int half   = lane >> 4;

    v8f acc;
    #pragma unroll
    for (int r = 0; r < 8; ++r) acc[r] = 0.0f;

    const unsigned short* hrow = h_bf + (size_t)(m_tile * 16 + n) * H_DIM;
    for (int k0 = 0; k0 < H_DIM; k0 += 32) {
        Frag32 a, b;
        const unsigned short* p1 = hrow + k0 + 8 * half;
        a.u[0] = *(const uint4*)(p1);
        a.u[1] = *(const uint4*)(p1 + 16);
        const unsigned short* wp = fcw + (size_t)(n0 + n) * H_DIM + k0 + 8 * half;
        b.u[0] = *(const uint4*)(wp);
        b.u[1] = *(const uint4*)(wp + 16);
        acc = __builtin_amdgcn_wmma_f32_16x16x32_bf16(
            false, a.bf, false, b.bf, (short)0, acc, false, false);
    }

    const int col = n0 + n;
    #pragma unroll
    for (int r = 0; r < 8; ++r) {
        const int m = m_tile * 16 + r + 8 * half;
        out[(size_t)m * H_DIM + col] = acc[r] + fcb[col];
    }
}

// ------------------------------- launcher ----------------------------------

extern "C" void kernel_launch(void* const* d_in, const int* in_sizes, int n_in,
                              void* d_out, int out_size, void* d_ws, size_t ws_size,
                              hipStream_t stream) {
    const float* x    = (const float*)d_in[0];
    const float* W_ih = (const float*)d_in[1];
    const float* W_hh = (const float*)d_in[2];
    const float* b_ih = (const float*)d_in[3];
    const float* b_hh = (const float*)d_in[4];
    const float* fc_w = (const float*)d_in[5];
    const float* fc_b = (const float*)d_in[6];
    float* out = (float*)d_out;

    // Workspace layout (bytes, all 256B-aligned)
    char* ws = (char*)d_ws;
    unsigned short* wih_bf = (unsigned short*)(ws);                    // 4 MB
    unsigned short* whh_bf = (unsigned short*)(ws + 4194304);          // 8 MB
    unsigned short* fcw_bf = (unsigned short*)(ws + 12582912);         // 2 MB
    float*          bsum   = (float*)(ws + 14680064);                  // 16 KB
    unsigned short* h0     = (unsigned short*)(ws + 14696448);         // 256 KB
    unsigned short* h1     = (unsigned short*)(ws + 14958592);         // 256 KB
    float*          cbuf   = (float*)(ws + 15220736);                  // 512 KB
    unsigned short* x_bf   = (unsigned short*)(ws + 15745024);         // 64 MB
    const size_t ws_needed_xbf = 15745024 + (size_t)BATCH * SEQ * D_IN * 2;
    const bool use_xbf = (ws_size >= ws_needed_xbf);

    // 1) weight conversion f32 -> bf16 (kept row-major: B[k][n]=W[n][k] is
    //    contiguous in k, so no transpose is needed for WMMA B fragments)
    k_f32_to_bf16_v8<<<1024, 256, 0, stream>>>(W_ih, (unsigned int*)wih_bf,
                                               G4 * D_IN / 8);
    k_f32_to_bf16_v8<<<2048, 256, 0, stream>>>(W_hh, (unsigned int*)whh_bf,
                                               G4 * H_DIM / 8);
    k_f32_to_bf16_v8<<<512,  256, 0, stream>>>(fc_w, (unsigned int*)fcw_bf,
                                               H_DIM * H_DIM / 8);
    if (use_xbf)   // one bandwidth-bound pass instead of 64x per-step converts
        k_f32_to_bf16_v8<<<8192, 256, 0, stream>>>(x, (unsigned int*)x_bf,
                                                   BATCH * SEQ * D_IN / 8);

    // 2) bias sum + zero state
    k_bias_zero<<<512, 256, 0, stream>>>(b_ih, b_hh, bsum, cbuf,
                                         (unsigned int*)h0, (unsigned int*)h1);

    // 3) 512 fused (x-proj + recurrent GEMM + cell) steps; h double-buffered
    for (int t = 0; t < SEQ; ++t) {
        const unsigned short* hin  = (t & 1) ? h1 : h0;
        unsigned short*       hout = (t & 1) ? h0 : h1;
        if (use_xbf)
            k_lstm_step<true><<<64, 256, 0, stream>>>(x, x_bf, wih_bf, whh_bf,
                                                      bsum, hin, hout, cbuf, t);
        else
            k_lstm_step<false><<<64, 256, 0, stream>>>(x, x_bf, wih_bf, whh_bf,
                                                       bsum, hin, hout, cbuf, t);
    }

    // 4) final FC (after t=511, latest h lives in h0)
    k_fc<<<64, 256, 0, stream>>>(h0, fcw_bf, fc_b, out);
}
